// MSN_15470472200294
// MI455X (gfx1250) — matrix operand order repaired and checked by reading, not compiled, not run on hardware
//
#include <hip/hip_runtime.h>
#include <math.h>

typedef __attribute__((ext_vector_type(16))) _Float16 v16h;
typedef __attribute__((ext_vector_type(8)))  float    v8f;

#define B_   4
#define K_   15
#define N_   2048
#define R_   16
#define NC_  16
#define PC_  128          // N_/NC_
#define C_   256
#define OUTC_ (4*C_ + K_) // 1039

// ---------------------------------------------------------------------------
// WMMA GEMM: Y[b][m][n] = sum_k W[m][k] * X[b][k][n] + bias[m]
// One wave per 16x64 output strip (NT=4 tiles), f16 inputs, f32 accumulate.
// Kd is a template parameter so all tail guards fold at compile time:
//  - Kd%32==0: A fragment = four aligned float4 loads per lane, B loads
//    unguarded (coalesced b32 across lanes 0..15).
//  - Kd=15: single k-step; clamp-address + select-zero => unconditional loads.
// A layout (16x32 f16): row M = lane&15; half h -> K = (h<8?h:h+8) + 8*(lane>>4)
// B layout (32x16 f16): col N = lane&15; half h -> K = (h<8?h:h+8) + 8*(lane>>4)
// C/D layout (16x16 f32): col N = lane&15; vgpr v -> M = v + 8*(lane>>4)
// ---------------------------------------------------------------------------
template<int Kd>
__global__ void msn_wmma_gemm(const float* __restrict__ W, const float* __restrict__ X,
                              const float* __restrict__ bias, float* __restrict__ Y,
                              int M, int N) {
  constexpr int NT = 4;
  const int lane = threadIdx.x & 31;
  const int n0 = blockIdx.x * (16 * NT);
  const int m0 = blockIdx.y * 16;
  const int b  = blockIdx.z;
  const int col = lane & 15;
  const int hi  = lane >> 4;
  const float* Xb   = X + (size_t)b * Kd * N;
  const float* Wrow = W + (size_t)(m0 + col) * Kd;

  v8f acc[NT];
#pragma unroll
  for (int t = 0; t < NT; ++t) acc[t] = (v8f){0.f,0.f,0.f,0.f,0.f,0.f,0.f,0.f};

  constexpr int KSTEPS = (Kd + 31) / 32;
#pragma unroll
  for (int ks = 0; ks < KSTEPS; ++ks) {
    const int kb = ks * 32;
    if (ks + 1 < KSTEPS)
      __builtin_prefetch(&Xb[(size_t)(kb + 32) * N + n0 + col], 0, 1);

    v16h a;
    if constexpr (Kd % 32 == 0) {
      // halves 0..7  = W[m][kb + hi*8 .. +7], halves 8..15 = W[m][kb+16+hi*8 .. +7]
      const float4 wa0 = *(const float4*)(Wrow + kb + hi * 8);
      const float4 wa1 = *(const float4*)(Wrow + kb + hi * 8 + 4);
      const float4 wb0 = *(const float4*)(Wrow + kb + 16 + hi * 8);
      const float4 wb1 = *(const float4*)(Wrow + kb + 16 + hi * 8 + 4);
      a[0]  = (_Float16)wa0.x; a[1]  = (_Float16)wa0.y;
      a[2]  = (_Float16)wa0.z; a[3]  = (_Float16)wa0.w;
      a[4]  = (_Float16)wa1.x; a[5]  = (_Float16)wa1.y;
      a[6]  = (_Float16)wa1.z; a[7]  = (_Float16)wa1.w;
      a[8]  = (_Float16)wb0.x; a[9]  = (_Float16)wb0.y;
      a[10] = (_Float16)wb0.z; a[11] = (_Float16)wb0.w;
      a[12] = (_Float16)wb1.x; a[13] = (_Float16)wb1.y;
      a[14] = (_Float16)wb1.z; a[15] = (_Float16)wb1.w;
    } else {
#pragma unroll
      for (int h = 0; h < 16; ++h) {
        int k  = kb + (h < 8 ? h : h + 8) + hi * 8;
        int kc = k < Kd ? k : Kd - 1;          // clamp: load stays unconditional
        float av = Wrow[kc];
        if (k >= Kd) av = 0.f;                 // v_cndmask, no branch
        a[h] = (_Float16)av;
      }
    }

#pragma unroll
    for (int t = 0; t < NT; ++t) {
      v16h bm;
#pragma unroll
      for (int h = 0; h < 16; ++h) {
        int k = kb + (h < 8 ? h : h + 8) + hi * 8;
        if constexpr (Kd % 32 == 0) {
          bm[h] = (_Float16)Xb[(size_t)k * N + n0 + t * 16 + col];
        } else {
          int kc = k < Kd ? k : Kd - 1;
          float bv = Xb[(size_t)kc * N + n0 + t * 16 + col];
          if (k >= Kd) bv = 0.f;
          bm[h] = (_Float16)bv;
        }
      }
      acc[t] = __builtin_amdgcn_wmma_f32_16x16x32_f16(false, a, false, bm,
                                                      (short)0, acc[t], false, false);
    }
  }

  float* Yb = Y + (size_t)b * M * N;
#pragma unroll
  for (int t = 0; t < NT; ++t) {
#pragma unroll
    for (int v = 0; v < 8; ++v) {
      int m = m0 + v + hi * 8;
      Yb[(size_t)m * N + n0 + t * 16 + col] = acc[t][v] + bias[m];
    }
  }
}

// ------------------- training-mode BatchNorm over (B, N) -------------------
__global__ void msn_bn_stats(const float* __restrict__ Y, float* __restrict__ stats,
                             int M, int N) {
  int m = blockIdx.x;
  float s = 0.f, sq = 0.f;
  for (int b = 0; b < B_; ++b) {
    const float* p = Y + ((size_t)b * M + m) * N;
    for (int n = threadIdx.x; n < N; n += blockDim.x) {
      float v = p[n];
      s += v; sq += v * v;
    }
  }
  __shared__ float rs[256], rq[256];
  rs[threadIdx.x] = s; rq[threadIdx.x] = sq;
  __syncthreads();
  for (int st = blockDim.x >> 1; st > 0; st >>= 1) {
    if ((int)threadIdx.x < st) {
      rs[threadIdx.x] += rs[threadIdx.x + st];
      rq[threadIdx.x] += rq[threadIdx.x + st];
    }
    __syncthreads();
  }
  if (threadIdx.x == 0) { stats[m] = rs[0]; stats[M + m] = rq[0]; }
}

__global__ void msn_bn_apply(float* __restrict__ Y, const float* __restrict__ stats,
                             const float* __restrict__ g, const float* __restrict__ bt,
                             int M, int N, int relu) {
  size_t i = (size_t)blockIdx.x * blockDim.x + threadIdx.x;
  size_t total = (size_t)B_ * M * N;
  if (i >= total) return;
  int m = (int)((i / (size_t)N) % (size_t)M);
  float cnt = (float)(B_ * N);
  float mean = stats[m] / cnt;
  float var  = stats[M + m] / cnt - mean * mean;
  float v = (Y[i] - mean) * rsqrtf(var + 1e-5f) * g[m] + bt[m];
  if (relu) v = fmaxf(v, 0.f);
  Y[i] = v;
}

// ------------------- max over points -------------------
__global__ void msn_maxpool_n(const float* __restrict__ Y, float* __restrict__ out, int N) {
  int bm = blockIdx.x;
  const float* p = Y + (size_t)bm * N;
  float mx = -1e30f;
  for (int n = threadIdx.x; n < N; n += blockDim.x) mx = fmaxf(mx, p[n]);
  __shared__ float r[256];
  r[threadIdx.x] = mx; __syncthreads();
  for (int st = blockDim.x >> 1; st > 0; st >>= 1) {
    if ((int)threadIdx.x < st) r[threadIdx.x] = fmaxf(r[threadIdx.x], r[threadIdx.x + st]);
    __syncthreads();
  }
  if (threadIdx.x == 0) out[bm] = r[0];
}

// ------------------- tiny FC layers (B=4) -------------------
__global__ void msn_fc(const float* __restrict__ in, const float* __restrict__ w,
                       const float* __restrict__ bias, float* __restrict__ out,
                       int In, int Out) {
  int o = blockIdx.x;
  float acc[B_] = {0.f, 0.f, 0.f, 0.f};
  for (int i = threadIdx.x; i < In; i += blockDim.x) {
    float wv = w[(size_t)o * In + i];
#pragma unroll
    for (int b = 0; b < B_; ++b) acc[b] += wv * in[(size_t)b * In + i];
  }
  __shared__ float r[128];
  for (int b = 0; b < B_; ++b) {
    r[threadIdx.x] = acc[b]; __syncthreads();
    for (int st = blockDim.x >> 1; st > 0; st >>= 1) {
      if ((int)threadIdx.x < st) r[threadIdx.x] += r[threadIdx.x + st];
      __syncthreads();
    }
    if (threadIdx.x == 0) out[(size_t)b * Out + o] = r[0] + bias[o];
    __syncthreads();
  }
}

__global__ void msn_bn_batch(float* __restrict__ x, const float* __restrict__ g,
                             const float* __restrict__ bt, int O, int relu) {
  int o = blockIdx.x * blockDim.x + threadIdx.x;
  if (o >= O) return;
  float v0 = x[o], v1 = x[O + o], v2 = x[2 * O + o], v3 = x[3 * O + o];
  float mean = (v0 + v1 + v2 + v3) * 0.25f;
  float d0 = v0 - mean, d1 = v1 - mean, d2 = v2 - mean, d3 = v3 - mean;
  float var = (d0 * d0 + d1 * d1 + d2 * d2 + d3 * d3) * 0.25f;
  float rs = rsqrtf(var + 1e-5f) * g[o];
  float bb = bt[o];
  float y0 = d0 * rs + bb, y1 = d1 * rs + bb, y2 = d2 * rs + bb, y3 = d3 * rs + bb;
  if (relu) {
    y0 = fmaxf(y0, 0.f); y1 = fmaxf(y1, 0.f);
    y2 = fmaxf(y2, 0.f); y3 = fmaxf(y3, 0.f);
  }
  x[o] = y0; x[O + o] = y1; x[2 * O + o] = y2; x[3 * O + o] = y3;
}

__global__ void msn_make_trans(const float* __restrict__ fc, float* __restrict__ tr) {
  int i = blockIdx.x * blockDim.x + threadIdx.x;
  if (i >= B_ * K_ * K_) return;
  int ij = i % (K_ * K_);
  tr[i] = fc[i] + (((ij / K_) == (ij % K_)) ? 1.f : 0.f);
}

// xt[b][k][n] = sum_c x[b][c][n] * trans[b][c][k]
__global__ void msn_apply_trans(const float* __restrict__ x, const float* __restrict__ tr,
                                float* __restrict__ xt) {
  int b = blockIdx.x;
  int n = blockIdx.y * blockDim.x + threadIdx.x;
  __shared__ float st[K_ * K_];
  for (int i = threadIdx.x; i < K_ * K_; i += blockDim.x) st[i] = tr[b * K_ * K_ + i];
  __syncthreads();
  float xv[K_];
#pragma unroll
  for (int c = 0; c < K_; ++c) xv[c] = x[((size_t)b * K_ + c) * N_ + n];
  for (int k = 0; k < K_; ++k) {
    float a = 0.f;
#pragma unroll
    for (int c = 0; c < K_; ++c) a += xv[c] * st[c * K_ + k];
    xt[((size_t)b * K_ + k) * N_ + n] = a;
  }
}

// ------------------- per-(b,r) descending bitonic argsort (2048) -------------------
__global__ void msn_sort_desc(const float* __restrict__ scores, int* __restrict__ idx) {
  __shared__ float key[N_];
  __shared__ int   val[N_];
  const float* s = scores + (size_t)blockIdx.x * N_;
  for (int i = threadIdx.x; i < N_; i += blockDim.x) { key[i] = s[i]; val[i] = i; }
  __syncthreads();
  for (int k = 2; k <= N_; k <<= 1) {
    for (int j = k >> 1; j > 0; j >>= 1) {
      for (int i = threadIdx.x; i < N_; i += blockDim.x) {
        int ixj = i ^ j;
        if (ixj > i) {
          bool up = (i & k) == 0;              // 'up' segments sort descending
          float ki = key[i], kj = key[ixj];
          bool sw = up ? (ki < kj) : (ki > kj);
          if (sw) {
            key[i] = kj; key[ixj] = ki;
            int t = val[i]; val[i] = val[ixj]; val[ixj] = t;
          }
        }
      }
      __syncthreads();
    }
  }
  for (int i = threadIdx.x; i < N_; i += blockDim.x)
    idx[(size_t)blockIdx.x * N_ + i] = val[i];
}

// ------------------- gather sp_cube into out[:,0:256] + cabin max -------------------
__global__ void msn_gather_cabins(const float* __restrict__ f, const int* __restrict__ idx,
                                  float* __restrict__ out, float* __restrict__ cab) {
  int br = blockIdx.x;            // b*R + r
  int c  = blockIdx.y;
  int b = br / R_, r = br % R_;
  const float* fr = f + ((size_t)b * C_ + c) * N_;
  const int* id = idx + (size_t)br * N_;
  float* orow = out + (((size_t)b * OUTC_ + c) * R_ + r) * N_;
  float* crow = cab + (((size_t)b * C_ + c) * R_ + r) * NC_;
  __shared__ float red[PC_];
  for (int ch = 0; ch < NC_; ++ch) {
    int n = ch * PC_ + threadIdx.x;
    float v = fr[id[n]];
    orow[n] = v;
    red[threadIdx.x] = v; __syncthreads();
    for (int st = PC_ >> 1; st > 0; st >>= 1) {
      if ((int)threadIdx.x < st) red[threadIdx.x] = fmaxf(red[threadIdx.x], red[threadIdx.x + st]);
      __syncthreads();
    }
    if (threadIdx.x == 0) crow[ch] = red[0];
    __syncthreads();
  }
}

// ------------------- small (1,KW) convs over the cabin width -------------------
__global__ void msn_conv_width(const float* __restrict__ in, const float* __restrict__ w,
                               const float* __restrict__ bias, float* __restrict__ out,
                               int Win, int KW, int Wout) {
  int br = blockIdx.x, x0 = blockIdx.y;
  int b = br / R_, r = br % R_;
  __shared__ float s_in[C_ * 16];
  for (int i = threadIdx.x; i < C_ * Win; i += blockDim.x) {
    int ci = i / Win, xx = i % Win;
    s_in[i] = in[(((size_t)b * C_ + ci) * R_ + r) * Win + xx];
  }
  __syncthreads();
  int co = threadIdx.x;
  float acc = bias[co];
  const float* wr = w + (size_t)co * C_ * KW;
  for (int ci = 0; ci < C_; ++ci) {
    const float* sp = &s_in[ci * Win + x0];
    const float* wp = &wr[ci * KW];
    for (int kw = 0; kw < KW; ++kw) acc += wp[kw] * sp[kw];
  }
  out[(((size_t)b * C_ + co) * R_ + r) * Wout + x0] = acc;
}

// ------------------- (R,1) station conv -------------------
__global__ void msn_station(const float* __restrict__ trains, const float* __restrict__ w,
                            const float* __restrict__ bias, float* __restrict__ out) {
  int b = blockIdx.x;
  __shared__ float s_tr[C_ * R_];
  for (int i = threadIdx.x; i < C_ * R_; i += blockDim.x)
    s_tr[i] = trains[(size_t)b * C_ * R_ + i];
  __syncthreads();
  int co = threadIdx.x;
  float acc = bias[co];
  const float* wr = w + (size_t)co * C_ * R_;
  for (int i = 0; i < C_ * R_; ++i) acc += wr[i] * s_tr[i];
  out[(size_t)b * C_ + co] = acc;
}

// ------------------- broadcast/gather fill for channels 256..1038 -------------------
__global__ void msn_fill_rest(const float* __restrict__ cab, const float* __restrict__ trains,
                              const float* __restrict__ station, const float* __restrict__ xt,
                              const int* __restrict__ idx, float* __restrict__ out) {
  size_t i = (size_t)blockIdx.x * blockDim.x + threadIdx.x;
  const size_t total = (size_t)B_ * (3 * C_ + K_) * R_ * N_;
  if (i >= total) return;
  int n = (int)(i & (size_t)(N_ - 1));
  int r = (int)((i >> 11) & (size_t)(R_ - 1));
  size_t rest = i >> 15;
  int ch = (int)(rest % (size_t)(3 * C_ + K_));
  int b  = (int)(rest / (size_t)(3 * C_ + K_));
  float v;
  if (ch < C_) {
    v = cab[(((size_t)b * C_ + ch) * R_ + r) * NC_ + (n >> 7)];
  } else if (ch < 2 * C_) {
    v = trains[((size_t)b * C_ + (ch - C_)) * R_ + r];
  } else if (ch < 3 * C_) {
    v = station[(size_t)b * C_ + (ch - 2 * C_)];
  } else {
    int k = ch - 3 * C_;
    int id = idx[((size_t)b * R_ + r) * N_ + n];
    v = xt[((size_t)b * K_ + k) * N_ + id];
  }
  out[(((size_t)b * OUTC_ + (C_ + ch)) * R_ + r) * N_ + n] = v;
}

// ===========================================================================
extern "C" void kernel_launch(void* const* d_in, const int* in_sizes, int n_in,
                              void* d_out, int out_size, void* d_ws, size_t ws_size,
                              hipStream_t stream) {
  (void)in_sizes; (void)n_in; (void)out_size; (void)ws_size;
  const float* x = (const float*)d_in[0];
  int p = 1;
  const float* s_c1_w = (const float*)d_in[p++]; const float* s_c1_b = (const float*)d_in[p++];
  const float* s_c2_w = (const float*)d_in[p++]; const float* s_c2_b = (const float*)d_in[p++];
  const float* s_c3_w = (const float*)d_in[p++]; const float* s_c3_b = (const float*)d_in[p++];
  const float* s_fc1_w = (const float*)d_in[p++]; const float* s_fc1_b = (const float*)d_in[p++];
  const float* s_fc2_w = (const float*)d_in[p++]; const float* s_fc2_b = (const float*)d_in[p++];
  const float* s_fc3_w = (const float*)d_in[p++]; const float* s_fc3_b = (const float*)d_in[p++];
  const float* s_bn1_g = (const float*)d_in[p++]; const float* s_bn1_b = (const float*)d_in[p++];
  const float* s_bn2_g = (const float*)d_in[p++]; const float* s_bn2_b = (const float*)d_in[p++];
  const float* s_bn3_g = (const float*)d_in[p++]; const float* s_bn3_b = (const float*)d_in[p++];
  const float* s_bn4_g = (const float*)d_in[p++]; const float* s_bn4_b = (const float*)d_in[p++];
  const float* s_bn5_g = (const float*)d_in[p++]; const float* s_bn5_b = (const float*)d_in[p++];
  const float* bn1_g = (const float*)d_in[p++]; const float* bn1_b = (const float*)d_in[p++];
  const float* bn2_g = (const float*)d_in[p++]; const float* bn2_b = (const float*)d_in[p++];
  const float* bn3_g = (const float*)d_in[p++]; const float* bn3_b = (const float*)d_in[p++];
  const float* c1_w = (const float*)d_in[p++]; const float* c1_b = (const float*)d_in[p++];
  const float* c2_w = (const float*)d_in[p++]; const float* c2_b = (const float*)d_in[p++];
  const float* c3_w = (const float*)d_in[p++]; const float* c3_b = (const float*)d_in[p++];
  const float* sorter_w = (const float*)d_in[p++]; const float* sorter_b = (const float*)d_in[p++];
  const float* t1_w = (const float*)d_in[p++]; const float* t1_b = (const float*)d_in[p++];
  const float* t2_w = (const float*)d_in[p++]; const float* t2_b = (const float*)d_in[p++];
  const float* t3_w = (const float*)d_in[p++]; const float* t3_b = (const float*)d_in[p++];
  const float* t4_w = (const float*)d_in[p++]; const float* t4_b = (const float*)d_in[p++];
  const float* t5_w = (const float*)d_in[p++]; const float* t5_b = (const float*)d_in[p++];

  float* ws = (float*)d_ws;
  size_t off = 0;
  auto A = [&](size_t n) { float* q = ws + off; off += n; return q; };
  float* f1     = A((size_t)B_ * 64 * N_);
  float* f2     = A((size_t)B_ * 128 * N_);
  float* f3     = A((size_t)B_ * 1024 * N_);
  float* pooled = A((size_t)B_ * 1024);
  float* fcA    = A((size_t)B_ * 512);
  float* fcB    = A((size_t)B_ * 256);
  float* fcC    = A((size_t)B_ * K_ * K_);
  float* trb    = A((size_t)B_ * K_ * K_);
  float* xt     = A((size_t)B_ * K_ * N_);
  float* fenc   = A((size_t)B_ * C_ * N_);
  float* scores = A((size_t)B_ * R_ * N_);
  int*   idxb   = (int*)A((size_t)B_ * R_ * N_);
  float* cab    = A((size_t)B_ * C_ * R_ * NC_);
  float* t1o    = A((size_t)B_ * C_ * R_ * 12);
  float* t2o    = A((size_t)B_ * C_ * R_ * 8);
  float* t3o    = A((size_t)B_ * C_ * R_ * 4);
  float* trn    = A((size_t)B_ * C_ * R_);
  float* stn    = A((size_t)B_ * C_);
  float* stats  = A(2 * 1024);
  float* e1 = f1;   // STN buffers are dead by the time the encoder runs
  float* e2 = f2;
  float* out = (float*)d_out;

  dim3 w32(32);
  const int GX = N_ / 64;   // 4 n-tiles per wave
  // ---- STN ----
  msn_wmma_gemm<K_><<<dim3(GX, 4, B_), w32, 0, stream>>>(s_c1_w, x, s_c1_b, f1, 64, N_);
  msn_bn_stats<<<64, 256, 0, stream>>>(f1, stats, 64, N_);
  msn_bn_apply<<<(B_*64*N_ + 255)/256, 256, 0, stream>>>(f1, stats, s_bn1_g, s_bn1_b, 64, N_, 1);
  msn_wmma_gemm<64><<<dim3(GX, 8, B_), w32, 0, stream>>>(s_c2_w, f1, s_c2_b, f2, 128, N_);
  msn_bn_stats<<<128, 256, 0, stream>>>(f2, stats, 128, N_);
  msn_bn_apply<<<(B_*128*N_ + 255)/256, 256, 0, stream>>>(f2, stats, s_bn2_g, s_bn2_b, 128, N_, 1);
  msn_wmma_gemm<128><<<dim3(GX, 64, B_), w32, 0, stream>>>(s_c3_w, f2, s_c3_b, f3, 1024, N_);
  msn_bn_stats<<<1024, 256, 0, stream>>>(f3, stats, 1024, N_);
  msn_bn_apply<<<(B_*1024*N_ + 255)/256, 256, 0, stream>>>(f3, stats, s_bn3_g, s_bn3_b, 1024, N_, 1);
  msn_maxpool_n<<<B_ * 1024, 256, 0, stream>>>(f3, pooled, N_);
  msn_fc<<<512, 128, 0, stream>>>(pooled, s_fc1_w, s_fc1_b, fcA, 1024, 512);
  msn_bn_batch<<<2, 256, 0, stream>>>(fcA, s_bn4_g, s_bn4_b, 512, 1);
  msn_fc<<<256, 128, 0, stream>>>(fcA, s_fc2_w, s_fc2_b, fcB, 512, 256);
  msn_bn_batch<<<1, 256, 0, stream>>>(fcB, s_bn5_g, s_bn5_b, 256, 1);
  msn_fc<<<K_ * K_, 128, 0, stream>>>(fcB, s_fc3_w, s_fc3_b, fcC, 256, K_ * K_);
  msn_make_trans<<<(B_*K_*K_ + 255)/256, 256, 0, stream>>>(fcC, trb);
  msn_apply_trans<<<dim3(B_, N_/128), 128, 0, stream>>>(x, trb, xt);

  // ---- encoder ----
  msn_wmma_gemm<K_><<<dim3(GX, 4, B_), w32, 0, stream>>>(c1_w, xt, c1_b, e1, 64, N_);
  msn_bn_stats<<<64, 256, 0, stream>>>(e1, stats, 64, N_);
  msn_bn_apply<<<(B_*64*N_ + 255)/256, 256, 0, stream>>>(e1, stats, bn1_g, bn1_b, 64, N_, 1);
  msn_wmma_gemm<64><<<dim3(GX, 8, B_), w32, 0, stream>>>(c2_w, e1, c2_b, e2, 128, N_);
  msn_bn_stats<<<128, 256, 0, stream>>>(e2, stats, 128, N_);
  msn_bn_apply<<<(B_*128*N_ + 255)/256, 256, 0, stream>>>(e2, stats, bn2_g, bn2_b, 128, N_, 1);
  msn_wmma_gemm<128><<<dim3(GX, 16, B_), w32, 0, stream>>>(c3_w, e2, c3_b, fenc, 256, N_);
  msn_bn_stats<<<256, 256, 0, stream>>>(fenc, stats, 256, N_);
  msn_bn_apply<<<(B_*256*N_ + 255)/256, 256, 0, stream>>>(fenc, stats, bn3_g, bn3_b, 256, N_, 0);

  // ---- sorter scores + argsort ----
  msn_wmma_gemm<C_><<<dim3(GX, 1, B_), w32, 0, stream>>>(sorter_w, fenc, sorter_b, scores, R_, N_);
  msn_sort_desc<<<B_ * R_, 512, 0, stream>>>(scores, idxb);

  // ---- sp_cube gather (writes out channels [0,256)) + cabin max ----
  msn_gather_cabins<<<dim3(B_ * R_, C_), PC_, 0, stream>>>(fenc, idxb, out, cab);

  // ---- trains / station ----
  msn_conv_width<<<dim3(B_ * R_, 12), 256, 0, stream>>>(cab, t1_w, t1_b, t1o, 16, 5, 12);
  msn_conv_width<<<dim3(B_ * R_,  8), 256, 0, stream>>>(t1o, t2_w, t2_b, t2o, 12, 5, 8);
  msn_conv_width<<<dim3(B_ * R_,  4), 256, 0, stream>>>(t2o, t3_w, t3_b, t3o, 8, 5, 4);
  msn_conv_width<<<dim3(B_ * R_,  1), 256, 0, stream>>>(t3o, t4_w, t4_b, trn, 4, 4, 1);
  msn_station<<<B_, 256, 0, stream>>>(trn, t5_w, t5_b, stn);

  // ---- broadcast/gather fill for channels [256,1039) ----
  size_t total = (size_t)B_ * (3 * C_ + K_) * R_ * N_;
  msn_fill_rest<<<(unsigned)((total + 255) / 256), 256, 0, stream>>>(cab, trn, stn, xt, idxb, out);
}
